// HyperGraphConv_2808908612025
// MI455X (gfx1250) — matrix operand order repaired and checked by reading, not compiled, not run on hardware
//
#include <hip/hip_runtime.h>

// ---------------- problem constants ----------------
#define B_    32
#define F_    64
#define T_    20
#define N_    512
#define A_    64
#define EINC  2048
#define NE    128     // NUM_EDGES
#define COUT  128
#define CIN   128

#define RS    132     // padded LDS row stride (elements) for 128-wide tiles -> bank spread

// ---------------- LDS layout (bytes), total 323584 <= 327680 (320KB/WGP) ----------------
#define OFF_FEAT   0          // bf16 [512][132] = 135168 ; later aliased as f32 sOut[256][128] = 131072
#define OFF_AGG    135168     // f32  [128][132] = 67584  (node->edge aggregated feat, f32 accum)
#define OFF_EDGE   202752     // f32  [128][132] = 67584  (edge_feat = Binv * agg @ theta^T)
#define OFF_THETA  270336     // bf16 [128][132] = 33792
#define OFF_NIDX   304128     // int  [2048]
#define OFF_EIDX   312320     // int  [2048]
#define OFF_DINV   320512     // f32  [512]  (deg counter, then 1/deg)
#define OFF_BINV   322560     // f32  [128]
#define OFF_BIAS   323072     // f32  [128]
#define SMEM_BYTES 323584

typedef __bf16 v16bf __attribute__((ext_vector_type(16)));
typedef float  v8f   __attribute__((ext_vector_type(8)));

__global__ void __launch_bounds__(1024, 1)
hypergraph_conv_kernel(const float* __restrict__ x,      // [B,F,T,N]
                       const int*   __restrict__ H,      // [B,T,2,EINC]
                       const float* __restrict__ ssa,    // [B,T,N,A]
                       const float* __restrict__ theta,  // [COUT,CIN]
                       const float* __restrict__ bias,   // [COUT]
                       float*       __restrict__ out)    // [B,T,N,COUT] flat
{
    extern __shared__ char smem[];
    __bf16* sFeat  = (__bf16*)(smem + OFF_FEAT);
    float*  sAgg   = (float*) (smem + OFF_AGG);
    float*  sEdgeF = (float*) (smem + OFF_EDGE);
    __bf16* sTh    = (__bf16*)(smem + OFF_THETA);
    int*    sNidx  = (int*)   (smem + OFF_NIDX);
    int*    sEidx  = (int*)   (smem + OFF_EIDX);
    float*  sDinv  = (float*) (smem + OFF_DINV);
    float*  sBinv  = (float*) (smem + OFF_BINV);
    float*  sBias  = (float*) (smem + OFF_BIAS);
    float*  sOut   = (float*) (smem + OFF_FEAT);   // alias: feat dead after GEMM phase

    const int tid  = threadIdx.x;          // 0..1023
    const int lane = tid & 31;
    const int wave = tid >> 5;             // 0..31
    const int bt   = blockIdx.x;           // 0..639
    const int b    = bt / T_;
    const int t    = bt % T_;

    const size_t hbase = (size_t)bt * 2 * EINC;
    __builtin_prefetch(&H[hbase], 0, 0);
    __builtin_prefetch(&theta[0], 0, 0);

    // ---------- Phase 0: stage inputs into LDS, zero accumulators ----------
    for (int e = tid; e < EINC; e += 1024) {
        sNidx[e] = H[hbase + e];           // node ids
        sEidx[e] = H[hbase + EINC + e];    // hyperedge ids
    }
    for (int i = tid; i < N_; i += 1024) sDinv[i] = 0.0f;
    if (tid < NE)   sBinv[tid] = 0.0f;
    if (tid < COUT) sBias[tid] = bias[tid];
    for (int i = tid; i < NE * RS; i += 1024) sAgg[i] = 0.0f;

    for (int i = tid; i < COUT * CIN; i += 1024) {     // theta -> bf16 LDS, row-major
        int r = i >> 7, k = i & 127;
        sTh[r * RS + k] = (__bf16)theta[i];
    }
    {   // feat[n][c] = c<64 ? x[b,c,t,n] : ssa[b,t,n,c-64]   (stored bf16)
        const size_t xbase = ((size_t)b * F_ * T_ + t) * N_;   // + c*T*N + n
        for (int i = tid; i < F_ * N_; i += 1024) {            // coalesced in n
            int n = i & (N_ - 1);
            int c = i >> 9;
            sFeat[n * RS + c] = (__bf16)x[xbase + (size_t)c * T_ * N_ + n];
        }
        const size_t sbase = (size_t)bt * N_ * A_;             // contiguous
        for (int i = tid; i < N_ * A_; i += 1024) {
            int c = i & (A_ - 1);
            int n = i >> 6;
            sFeat[n * RS + F_ + c] = (__bf16)ssa[sbase + i];
        }
    }
    __syncthreads();

    // ---------- Phase 1: degree counts (LDS float atomics) ----------
    for (int e = tid; e < EINC; e += 1024) {
        atomicAdd(&sDinv[sNidx[e]], 1.0f);
        atomicAdd(&sBinv[sEidx[e]], 1.0f);
    }
    __syncthreads();

    // ---------- Phase 2: invert degrees + node->edge feat aggregation ----------
    if (tid < N_) { float v = sDinv[tid]; sDinv[tid] = v > 0.0f ? 1.0f / v : 0.0f; }
    if (tid < NE) { float v = sBinv[tid]; sBinv[tid] = v > 0.0f ? 1.0f / v : 0.0f; }
    // GEMM commutes with this aggregation: agg[e] = sum feat[node] ; edge_feat = Binv*(agg @ theta^T)
    for (int e = wave; e < EINC; e += 32) {            // one incidence per wave
        const __bf16* frow = sFeat + sNidx[e] * RS;
        float*        arow = sAgg  + sEidx[e] * RS;
        #pragma unroll
        for (int cc = 0; cc < 4; ++cc) {
            int c = lane + cc * 32;
            atomicAdd(&arow[c], (float)frow[c]);
        }
    }
    __syncthreads();

    // ---------- Phase 3: edge_feat = Binv * (agg[128x128] @ theta^T[128x128]) via bf16 WMMA ----------
    {
        const int half = lane >> 4;     // 0: lanes 0-15, 1: lanes 16-31
        const int l16  = lane & 15;
        #pragma unroll
        for (int s = 0; s < 2; ++s) {
            const int tile = wave * 2 + s;           // 64 tiles of 16x16, 2 per wave
            const int mi = tile >> 3;                // edge-row block
            const int nj = tile & 7;                 // out-channel block
            const float*  arow = sAgg + (mi * 16 + l16) * RS;  // A row (M = l16)
            const __bf16* brow = sTh  + (nj * 16 + l16) * RS;  // theta row = output column
            v8f acc = {};
            #pragma unroll
            for (int kk = 0; kk < 4; ++kk) {
                v16bf afrag, bfrag;
                // A 16x32 bf16 layout: lanes 0-15 K=0-15; lanes 16-31 K=16-31 (contiguous)
                const int ka = kk * 32 + 16 * half;
                #pragma unroll
                for (int j = 0; j < 16; ++j) {
                    afrag[j]     = (__bf16)arow[ka + j];
                }
                // B 32x16 bf16 layout: lanes 0-15 K=0-15; lanes 16-31 K=16-31 (contiguous)
                const int kb = kk * 32 + 16 * half;
                #pragma unroll
                for (int j = 0; j < 16; ++j) bfrag[j] = brow[kb + j];
                acc = __builtin_amdgcn_wmma_f32_16x16x32_bf16(
                          false, afrag, false, bfrag, (short)0, acc, false, false);
            }
            // C/D layout: VGPR r -> M = r + 8*half, N = l16 ; fold Binv scale here
            #pragma unroll
            for (int r = 0; r < 8; ++r) {
                const int e = mi * 16 + r + 8 * half;
                sEdgeF[e * RS + nj * 16 + l16] = acc[r] * sBinv[e];
            }
        }
    }
    __syncthreads();

    // ---------- Phase 4: edge->node scatter with Dinv folded per-contribution, 2 chunks of 256 rows ----------
    float* outbase = out + (size_t)bt * N_ * COUT;
    for (int chunk = 0; chunk < 2; ++chunk) {
        for (int i = tid; i < 256 * COUT; i += 1024) sOut[i] = 0.0f;
        __syncthreads();
        for (int e = wave; e < EINC; e += 32) {
            const int n = sNidx[e];
            if ((n >> 8) == chunk) {
                const float  dsc  = sDinv[n];
                const float* erow = sEdgeF + sEidx[e] * RS;
                float*       orow = sOut + (n & 255) * COUT;
                #pragma unroll
                for (int cc = 0; cc < 4; ++cc) {
                    int c = lane + cc * 32;
                    atomicAdd(&orow[c], dsc * erow[c]);
                }
            }
        }
        __syncthreads();
        for (int i = tid; i < 256 * COUT; i += 1024) {      // coalesced writeback + bias
            int c = i & (COUT - 1);
            outbase[(size_t)chunk * 256 * COUT + i] = sOut[i] + sBias[c];
        }
        __syncthreads();
    }
}

extern "C" void kernel_launch(void* const* d_in, const int* in_sizes, int n_in,
                              void* d_out, int out_size, void* d_ws, size_t ws_size,
                              hipStream_t stream) {
    (void)in_sizes; (void)n_in; (void)out_size; (void)d_ws; (void)ws_size;
    const float* x     = (const float*)d_in[0];
    const int*   H     = (const int*)  d_in[1];
    const float* ssa   = (const float*)d_in[2];
    /* d_in[3] = W : unused by the conv (matches reference) */
    const float* theta = (const float*)d_in[4];
    const float* bias  = (const float*)d_in[5];
    float*       out   = (float*)d_out;

    hipLaunchKernelGGL(hypergraph_conv_kernel,
                       dim3(B_ * T_), dim3(1024), SMEM_BYTES, stream,
                       x, H, ssa, theta, bias, out);
}
